// DirectConv2D_51110110822421
// MI455X (gfx1250) — 8-bit matrix operand order repaired and checked by reading, compiled, not run on hardware
//
#include <hip/hip_runtime.h>
#include <stdint.h>

typedef __attribute__((ext_vector_type(4)))  unsigned int u32x4;
typedef __attribute__((ext_vector_type(4)))  int          i32x4;
typedef __attribute__((ext_vector_type(16))) int          v16i;
typedef __attribute__((ext_vector_type(8)))  float        v8f;

typedef int v4i_gcc __attribute__((vector_size(16)));

#define FP8_LUT 0x4E4C4A4844403800ULL

__device__ __forceinline__ uint8_t fp8_of_small(int v) {
  return (uint8_t)((FP8_LUT >> (v * 8)) & 0xFF);
}

#if __has_builtin(__builtin_amdgcn_global_load_async_to_lds_b128) && \
    __has_builtin(__builtin_amdgcn_s_wait_asynccnt)
#define USE_ASYNC_LDS 1
#else
#define USE_ASYNC_LDS 0
#endif

__global__ void __launch_bounds__(256)
pack_x_nhwc_fp8(const int* __restrict__ x, uint8_t* __restrict__ xq) {
  __shared__ __align__(16) uint8_t tile[64 * 80];
  const int t = threadIdx.x;
  const int n = blockIdx.z, ct = blockIdx.y, hwt = blockIdx.x;
  {
    const int c_local  = t >> 2;
    const int hw_chunk = (t & 3) * 16;
    const int c   = ct * 64 + c_local;
    const int hw0 = hwt * 64 + hw_chunk;
    const int* src = x + ((size_t)(n * 256 + c) * 3136 + hw0);
#pragma unroll
    for (int kq = 0; kq < 4; ++kq) {
      i32x4 a = *(const i32x4*)(src + kq * 4);
#pragma unroll
      for (int e = 0; e < 4; ++e) {
        int v = a[e];
        v = v < 0 ? 0 : (v > 7 ? 7 : v);
        tile[(hw_chunk + kq * 4 + e) * 80 + c_local] = fp8_of_small(v);
      }
    }
  }
  __syncthreads();
  {
    const int hw_local = t >> 2;
    const int c_chunk  = (t & 3) * 16;
    u32x4 v = *(const u32x4*)(tile + hw_local * 80 + c_chunk);
    *(u32x4*)(xq + ((size_t)(n * 3136 + hwt * 64 + hw_local) * 256) + ct * 64 + c_chunk) = v;
  }
}

__global__ void __launch_bounds__(256)
pack_w_fp8(const float* __restrict__ w, uint8_t* __restrict__ wpk) {
  const int d = blockIdx.x * blockDim.x + threadIdx.x;
  if (d >= 147456) return;
  const int tile = d >> 9;
  const int rr   = d & 511;
  const int lane = rr >> 4;
  const int vv   = rr & 15;
  const int kstep = tile >> 4;
  const int ktile = tile & 15;
  const int rs = kstep >> 1;
  const int cb = kstep & 1;
  const int r3 = rs / 3, s3 = rs % 3;
  const int kout = ktile * 16 + (lane & 15);
  const int kbase = (vv >> 2) * 32 + (lane >> 4) * 16 + (vv & 3) * 4;
  unsigned int pack = 0;
#pragma unroll
  for (int j = 0; j < 4; ++j) {
    const int c = cb * 128 + kbase + j;
    const int widx = ((kout * 256 + c) * 3 + r3) * 3 + s3;
    int v = (int)w[widx];
    v = v < 0 ? 0 : (v > 7 ? 7 : v);
    pack |= ((unsigned int)fp8_of_small(v)) << (j * 8);
  }
  ((unsigned int*)wpk)[d] = pack;
}

__global__ void __launch_bounds__(256)
conv_fp8_wmma(const uint8_t* __restrict__ xq, const uint8_t* __restrict__ wpk,
              float* __restrict__ out) {
  __shared__ __align__(16) uint8_t smem[2][64 * 144];
  const int t    = threadIdx.x;
  const int lane = t & 31;
  const int wv   = t >> 5;
  const int wm   = wv >> 1;
  const int wn   = wv & 1;
  const int n    = blockIdx.z;
  const int mb   = blockIdx.y;
  const int p0   = blockIdx.x * 64;

  int sp = p0 + (t >> 2);
  sp = sp < 2915 ? sp : 2915;
  const int sh = sp / 54;
  const int sw = sp - sh * 54;
  const uint8_t* xsrc = xq + ((size_t)((n * 56 + sh) * 56 + sw) * 256) + (t & 3) * 32;
  const int sdst = (t >> 2) * 144 + (t & 3) * 32;

  v8f acc[2][2] = {};

#if USE_ASYNC_LDS
  typedef __attribute__((address_space(1))) v4i_gcc* gvec_t;
  typedef __attribute__((address_space(3))) v4i_gcc* lvec_t;
  auto stage = [&](int ks, int buf) {
    const int rs = ks >> 1, cb = ks & 1;
    const uint8_t* src = xsrc + ((rs / 3) * 56 + (rs % 3)) * 256 + cb * 128;
    uint8_t* dst = &smem[buf][sdst];
    gvec_t g = (gvec_t)(uintptr_t)src;
    lvec_t l = (lvec_t)(uint32_t)(uintptr_t)dst;
    __builtin_amdgcn_global_load_async_to_lds_b128(g, l, 0, 0);
    __builtin_amdgcn_global_load_async_to_lds_b128(g + 1, l + 1, 0, 0);
  };
#else
  auto stage = [&](int ks, int buf) {
    const int rs = ks >> 1, cb = ks & 1;
    const uint8_t* src = xsrc + ((rs / 3) * 56 + (rs % 3)) * 256 + cb * 128;
    u32x4 q0 = *(const u32x4*)(src);
    u32x4 q1 = *(const u32x4*)(src + 16);
    uint8_t* dst = &smem[buf][sdst];
    *(u32x4*)(dst)      = q0;
    *(u32x4*)(dst + 16) = q1;
  };
#endif

  stage(0, 0);
#if USE_ASYNC_LDS
  __builtin_amdgcn_s_wait_asynccnt(0);
#endif
  __syncthreads();

  const int brow = (wn * 32 + (lane & 15)) * 144 + (lane >> 4) * 16;

  for (int ks = 0; ks < 18; ++ks) {
    const int buf = ks & 1;
    if (ks + 1 < 18) stage(ks + 1, buf ^ 1);
    if (ks + 2 < 18)
      __builtin_prefetch(wpk + ((size_t)((ks + 2) * 16 + mb * 8 + wm * 2) * 32 + lane) * 64, 0, 3);

    v16i afrag[2], bfrag[2];
#pragma unroll
    for (int i = 0; i < 2; ++i) {
      const uint8_t* ap =
          wpk + ((size_t)(ks * 16 + mb * 8 + wm * 2 + i) * 32 + lane) * 64;
      union { u32x4 q[4]; v16i v; } ua;
#pragma unroll
      for (int q = 0; q < 4; ++q) ua.q[q] = *(const u32x4*)(ap + q * 16);
      afrag[i] = ua.v;
    }
#pragma unroll
    for (int j = 0; j < 2; ++j) {
      const uint8_t* bp = &smem[buf][brow + j * 16 * 144];
      union { u32x4 q[4]; v16i v; } ub;
#pragma unroll
      for (int q = 0; q < 4; ++q) ub.q[q] = *(const u32x4*)(bp + q * 32);
      bfrag[j] = ub.v;
    }
#pragma unroll
    for (int i = 0; i < 2; ++i)
#pragma unroll
      for (int j = 0; j < 2; ++j)
        acc[i][j] = __builtin_amdgcn_wmma_f32_16x16x128_fp8_fp8(
            afrag[i], bfrag[j], (short)0, acc[i][j], false, false);

#if USE_ASYNC_LDS
    __builtin_amdgcn_s_wait_asynccnt(0);
#endif
    __syncthreads();
  }

  const int pcol  = lane & 15;
  const int mhalf = (lane >> 4) * 8;
#pragma unroll
  for (int j = 0; j < 2; ++j) {
    const int p = p0 + (wn * 2 + j) * 16 + pcol;
    if (p < 2916) {
#pragma unroll
      for (int i = 0; i < 2; ++i) {
        const int k0 = (mb * 8 + wm * 2 + i) * 16 + mhalf;
#pragma unroll
        for (int vg = 0; vg < 8; ++vg)
          out[((size_t)(n * 256 + k0 + vg)) * 2916 + p] = acc[i][j][vg];
      }
    }
  }
}

extern "C" void kernel_launch(void* const* d_in, const int* in_sizes, int n_in,
                              void* d_out, int out_size, void* d_ws, size_t ws_size,
                              hipStream_t stream) {
  const int*   x = (const int*)d_in[0];
  const float* w = (const float*)d_in[1];
  float* out = (float*)d_out;

  uint8_t* xq  = (uint8_t*)d_ws;
  uint8_t* wpk = xq + 25690112;

  pack_x_nhwc_fp8<<<dim3(49, 4, 32), 256, 0, stream>>>(x, xq);
  pack_w_fp8<<<dim3(576, 1, 1), 256, 0, stream>>>(w, wpk);
  conv_fp8_wmma<<<dim3(46, 2, 32), 256, 0, stream>>>(xq, wpk, out);
}
